// RealNVP_45406394253437
// MI455X (gfx1250) — matrix operand order repaired and checked by reading, not compiled, not run on hardware
//
#include <hip/hip_runtime.h>
#include <hip/hip_bf16.h>

typedef __attribute__((ext_vector_type(16))) _Float16 v16h;
typedef __attribute__((ext_vector_type(8)))  float    v8f;

// ---------------------------------------------------------------------------
// Workspace layout (bytes), built once by prep kernel:
//   W1H [6][2][32][32] f16  : layer-1 weight row k(i), pre-swizzled per lane
//                             into A-fragment channel order (packed pairs)
//   B1H [6][2][32][32] f16  : layer-1 bias, same swizzle
//   W2F [6][2][4][2][32][16] f16 : w2 in B-fragment order
//                             (ntile, ktile, lane, 16 halfs contiguous = 32B)
//   B2C [6][2][64] f32      : layer-2 bias
//   W3C [6][2][64] f32      : w3 column j(i)
//   B3C [6][2] f32          : b3[j(i)]
// ---------------------------------------------------------------------------
#define OFF_W1H 0
#define OFF_B1H 24576
#define OFF_W2F 49152
#define OFF_B2C 147456
#define OFF_W3C 150528
#define OFF_B3C 153600

// A-fragment channel map (16-bit A 16x32, ISA 7.12.2):
// lane<16 : e=0..7 -> K=e,     e=8..15 -> K=e+8   (K 0..7,16..23)
// lane>=16: e=0..7 -> K=8+e,   e=8..15 -> K=16+e  (K 8..15,24..31)
__device__ __host__ inline int chanA(int lane, int kt, int e) {
    return 32 * kt + ((lane >> 4) << 3) + (e < 8 ? e : e + 8);
}

__global__ void nvp_prep(const float* __restrict__ sw1, const float* __restrict__ sb1,
                         const float* __restrict__ sw2, const float* __restrict__ sb2,
                         const float* __restrict__ sw3, const float* __restrict__ sb3,
                         const float* __restrict__ tw1, const float* __restrict__ tb1,
                         const float* __restrict__ tw2, const float* __restrict__ tb2,
                         const float* __restrict__ tw3, const float* __restrict__ tb3,
                         char* __restrict__ ws)
{
    _Float16* W1H = (_Float16*)(ws + OFF_W1H);
    _Float16* B1H = (_Float16*)(ws + OFF_B1H);
    _Float16* W2F = (_Float16*)(ws + OFF_W2F);
    float*    B2C = (float*)(ws + OFF_B2C);
    float*    W3C = (float*)(ws + OFF_W3C);
    float*    B3C = (float*)(ws + OFF_B3C);

    const int tid = blockIdx.x * blockDim.x + threadIdx.x;
    const int nth = gridDim.x * blockDim.x;

    // Layer-1 swizzle: W1H/B1H [i][m][lane][t], t = kt*16 + e
    for (int idx = tid; idx < 6 * 2 * 32 * 32; idx += nth) {
        int t = idx & 31, lane = (idx >> 5) & 31, m = (idx >> 10) & 1, i = idx >> 11;
        int k = (i & 1) ? 0 : 1;              // kept (input) coordinate
        int c = chanA(lane, t >> 4, t & 15);
        const float* w1 = m ? tw1 : sw1;      // [6][2][64]
        const float* b1 = m ? tb1 : sb1;      // [6][64]
        W1H[idx] = (_Float16)w1[(i * 2 + k) * 64 + c];
        B1H[idx] = (_Float16)b1[i * 64 + c];
    }
    // w2 -> B fragments: [i][m][nt][kt][lane][e]; k = 32*kt + 8*(lane/16) + (e<8 ? e : e+8),
    // n = nt*16 + lane%16  (row-striped-across-lanes B layout)
    for (int idx = tid; idx < 6 * 2 * 4 * 2 * 32 * 16; idx += nth) {
        int e = idx & 15, lane = (idx >> 4) & 31, kt = (idx >> 9) & 1;
        int nt = (idx >> 10) & 3, m = (idx >> 12) & 1, i = idx >> 13;
        int kk = 32 * kt + 8 * (lane >> 4) + (e < 8 ? e : e + 8);
        int nn = nt * 16 + (lane & 15);
        const float* w2 = m ? tw2 : sw2;      // [6][64][64]
        W2F[idx] = (_Float16)w2[i * 4096 + kk * 64 + nn];
    }
    // b2 and w3 column j
    for (int idx = tid; idx < 6 * 2 * 64; idx += nth) {
        int c = idx & 63, m = (idx >> 6) & 1, i = idx >> 7;
        const float* b2 = m ? tb2 : sb2;      // [6][64]
        const float* w3 = m ? tw3 : sw3;      // [6][64][2]
        int j = (i & 1) ? 1 : 0;              // updated coordinate
        B2C[idx] = b2[i * 64 + c];
        W3C[idx] = w3[i * 128 + c * 2 + j];
    }
    for (int idx = tid; idx < 12; idx += nth) {
        int m = idx & 1, i = idx >> 1;
        const float* b3 = m ? tb3 : sb3;      // [6][2]
        int j = (i & 1) ? 1 : 0;
        B3C[idx] = b3[i * 2 + j];
    }
}

// ---------------------------------------------------------------------------
// Main kernel: one wave32 owns a 16-point M-tile. z state duplicated across
// the two 16-lane halves so all lanes contribute to layer-1/layer-3 VALU work.
// Layer-1 runs as packed f16 vector math (v_pk_fma/v_pk_mul/v_pk_max)
// straight into the WMMA A fragments.
// ---------------------------------------------------------------------------
__global__ void __launch_bounds__(256) nvp_kernel(
    const float* __restrict__ x, const char* __restrict__ ws,
    float* __restrict__ out, int ntiles)
{
    __shared__ float oBuf[8][16];             // per-wave scalar bounce buffer

    const _Float16* W1H = (const _Float16*)(ws + OFF_W1H);
    const _Float16* B1H = (const _Float16*)(ws + OFF_B1H);
    const v16h*     W2F = (const v16h*)(ws + OFF_W2F);   // 32B-aligned chunks
    const float*    B2C = (const float*)(ws + OFF_B2C);
    const float*    W3C = (const float*)(ws + OFF_W3C);
    const float*    B3C = (const float*)(ws + OFF_B3C);

    const int lane  = threadIdx.x & 31;
    const int wslot = threadIdx.x >> 5;
    const int gw    = blockIdx.x * (blockDim.x >> 5) + wslot;
    const int nW    = gridDim.x * (blockDim.x >> 5);
    const int ln15  = lane & 15;
    const int hi    = lane >> 4;

    const _Float16 slope = (_Float16)0.01f;

    for (int tile = gw; tile < ntiles; tile += nW) {
        const int p = tile * 16 + ln15;
        float z0 = x[2 * p], z1 = x[2 * p + 1];
        float logdet = 0.f;

        for (int it = 0; it < 6; ++it) {      // i = 5..0
            const int  i    = 5 - it;
            const bool even = (i & 1) == 0;   // even: input z1, update z0
            const float zk  = even ? z1 : z0;
            float sv = 0.f, tv = 0.f;

            #pragma unroll
            for (int m = 0; m < 2; ++m) {     // 0 = s-net, 1 = t-net
                const int im = i * 2 + m;
                const v16h* w1v = (const v16h*)(W1H + im * 1024 + lane * 32);
                const v16h* b1v = (const v16h*)(B1H + im * 1024 + lane * 32);

                // Layer 1 (packed f16 vector math) -> A fragments (16x32 f16)
                const _Float16 zkh = (_Float16)zk;
                v16h a0 = w1v[0] * zkh + b1v[0];
                a0 = __builtin_elementwise_max(a0, a0 * slope);  // leaky relu
                v16h a1 = w1v[1] * zkh + b1v[1];
                a1 = __builtin_elementwise_max(a1, a1 * slope);

                // Hidden 64x64 GEMM on WMMA + fused bias/lrelu + w3 dot
                float partial[8];
                #pragma unroll
                for (int r = 0; r < 8; ++r) partial[r] = 0.f;

                #pragma unroll
                for (int nt = 0; nt < 4; ++nt) {
                    v16h bf0 = W2F[((im * 4 + nt) * 2 + 0) * 32 + lane];
                    v16h bf1 = W2F[((im * 4 + nt) * 2 + 1) * 32 + lane];
                    v8f c = {};
                    c = __builtin_amdgcn_wmma_f32_16x16x32_f16(
                            false, a0, false, bf0, (short)0, c, false, false);
                    c = __builtin_amdgcn_wmma_f32_16x16x32_f16(
                            false, a1, false, bf1, (short)0, c, false, false);
                    const float bias = B2C[im * 64 + nt * 16 + ln15];
                    const float w3v  = W3C[im * 64 + nt * 16 + ln15];
                    #pragma unroll
                    for (int r = 0; r < 8; ++r) {
                        float h = c[r] + bias;
                        h = fmaxf(h, 0.01f * h);       // branch-free leaky relu
                        partial[r] = fmaf(h, w3v, partial[r]);
                    }
                }

                // Reduce across the 16 lanes of each half (C layout: lane=N col)
                #pragma unroll
                for (int msk = 1; msk <= 8; msk <<= 1) {
                    #pragma unroll
                    for (int r = 0; r < 8; ++r)
                        partial[r] += __shfl_xor(partial[r], msk, 32);
                }
                // partial[r] now holds o for point M = r + 8*hi; bounce via LDS
                const float b3v = B3C[im];
                if (ln15 == 0) {
                    #pragma unroll
                    for (int r = 0; r < 8; ++r)
                        oBuf[wslot][hi * 8 + r] = partial[r] + b3v;
                }
                const float o = oBuf[wslot][ln15];   // same-wave DS, in order
                if (m == 0) sv = tanhf(o); else tv = o;
            }

            const float ex = __expf(-sv);
            if (even) z0 = (z0 - tv) * ex;
            else      z1 = (z1 - tv) * ex;
            logdet -= sv;
        }

        if (lane < 16) {
            // prior logp (std 2-D gaussian) + log-det
            out[tile * 16 + ln15] =
                -0.5f * (z0 * z0 + z1 * z1) - 1.8378770664093453f + logdet;
        }
    }
}

extern "C" void kernel_launch(void* const* d_in, const int* in_sizes, int n_in,
                              void* d_out, int out_size, void* d_ws, size_t ws_size,
                              hipStream_t stream) {
    const float* x   = (const float*)d_in[0];
    const float* sw1 = (const float*)d_in[1];
    const float* sb1 = (const float*)d_in[2];
    const float* sw2 = (const float*)d_in[3];
    const float* sb2 = (const float*)d_in[4];
    const float* sw3 = (const float*)d_in[5];
    const float* sb3 = (const float*)d_in[6];
    const float* tw1 = (const float*)d_in[7];
    const float* tb1 = (const float*)d_in[8];
    const float* tw2 = (const float*)d_in[9];
    const float* tb2 = (const float*)d_in[10];
    const float* tw3 = (const float*)d_in[11];
    const float* tb3 = (const float*)d_in[12];
    char* ws = (char*)d_ws;

    nvp_prep<<<64, 256, 0, stream>>>(sw1, sb1, sw2, sb2, sw3, sb3,
                                     tw1, tb1, tw2, tb2, tw3, tb3, ws);

    const int npoints = in_sizes[0] / 2;          // x is [N,2]
    const int ntiles  = npoints / 16;             // 16 points per wave
    int blocks = (ntiles + 7) / 8;                // 8 waves / 256-thread block
    if (blocks > 8192) blocks = 8192;
    nvp_kernel<<<blocks, 256, 0, stream>>>(x, ws, (float*)d_out, ntiles);
}
